// GAT_85229331022442
// MI455X (gfx1250) — matrix operand order repaired and checked by reading, not compiled, not run on hardware
//
#include <hip/hip_runtime.h>
#include <math.h>

// ---------------- CDNA5 WMMA types ----------------
typedef __attribute__((ext_vector_type(16))) __bf16 v16bf;
typedef __attribute__((ext_vector_type(8)))  float  v8f;

#define NEG_SLOPE 0.2f
#define KEY_NEG_INF 0x007FFFFFu   // monotone-uint key of -inf

// monotone float <-> uint mapping so integer atomicMax == float max
__device__ __forceinline__ unsigned f2key(float f) {
  int i = __float_as_int(f);
  return (unsigned)(i ^ ((i >> 31) | 0x80000000));
}
__device__ __forceinline__ float key2f(unsigned u) {
  int i = (u & 0x80000000u) ? (int)(u ^ 0x80000000u) : ~(int)u;
  return __int_as_float(i);
}

__device__ __forceinline__ float lrelu(float t) {
  return t > 0.f ? t : NEG_SLOPE * t;
}

// Build a 16x32 bf16 A-fragment for one lane from an f32 row pointer.
// Lane layout: lane = M + 16*khalf; element e -> K=b+e,
// with b = kk + khalf*16 folded into `xrow`.
__device__ __forceinline__ v16bf make_afrag(const float* __restrict__ xrow) {
  float4 a0 = *(const float4*)(xrow);
  float4 a1 = *(const float4*)(xrow + 4);
  float4 a2 = *(const float4*)(xrow + 8);
  float4 a3 = *(const float4*)(xrow + 12);
  v16bf af;
  af[0]=(__bf16)a0.x; af[1]=(__bf16)a0.y; af[2]=(__bf16)a0.z; af[3]=(__bf16)a0.w;
  af[4]=(__bf16)a1.x; af[5]=(__bf16)a1.y; af[6]=(__bf16)a1.z; af[7]=(__bf16)a1.w;
  af[8]=(__bf16)a2.x; af[9]=(__bf16)a2.y; af[10]=(__bf16)a2.z; af[11]=(__bf16)a2.w;
  af[12]=(__bf16)a3.x; af[13]=(__bf16)a3.y; af[14]=(__bf16)a3.z; af[15]=(__bf16)a3.w;
  return af;
}

// ---------------- W1 convert + transpose: w1t[c*512 + k] = bf16(W1[k*64 + c]) ----------------
__global__ void k_conv_w1(const float* __restrict__ W1, __bf16* __restrict__ w1t) {
  int tid = blockIdx.x * blockDim.x + threadIdx.x;
  if (tid >= 512 * 64) return;
  int c = tid >> 9;        // 0..63
  int k = tid & 511;       // 0..511
  w1t[tid] = (__bf16)W1[k * 64 + c];
}

// ---------------- GEMM1: h1[N,64] = x[N,512] @ W1[512,64]  (bf16 WMMA, f32 acc) ----------------
// block = 256 threads = 8 waves; each wave owns a 32-row strip (2 M-tiles) x full
// 64-col width (4 N-tiles) -> each B fragment feeds 2 WMMAs.
__global__ void k_gemm1(const float* __restrict__ x,
                        const __bf16* __restrict__ w1t,
                        float* __restrict__ h1, int n) {
  const int wave  = threadIdx.x >> 5;
  const int lane  = threadIdx.x & 31;
  const int m0    = (blockIdx.x * 8 + wave) * 32;
  if (m0 >= n) return;                              // uniform per wave

  const int lrow  = lane & 15;
  const int khalf = lane >> 4;                      // 0 or 1
  int row0 = m0 + lrow;        if (row0 >= n) row0 = n - 1;  // clamp (n%32==0 here)
  int row1 = m0 + 16 + lrow;   if (row1 >= n) row1 = n - 1;

  const float*  xr0 = x + (size_t)row0 * 512 + khalf * 16;
  const float*  xr1 = x + (size_t)row1 * 512 + khalf * 16;
  // B fragment base: lane holds col N = t*16 + lrow, K = kk + khalf*16 + e (contig 32B)
  const __bf16* wb  = w1t + (size_t)lrow * 512 + khalf * 16;

  v8f acc0[4] = {{0.f,0.f,0.f,0.f,0.f,0.f,0.f,0.f},
                 {0.f,0.f,0.f,0.f,0.f,0.f,0.f,0.f},
                 {0.f,0.f,0.f,0.f,0.f,0.f,0.f,0.f},
                 {0.f,0.f,0.f,0.f,0.f,0.f,0.f,0.f}};
  v8f acc1[4] = {{0.f,0.f,0.f,0.f,0.f,0.f,0.f,0.f},
                 {0.f,0.f,0.f,0.f,0.f,0.f,0.f,0.f},
                 {0.f,0.f,0.f,0.f,0.f,0.f,0.f,0.f},
                 {0.f,0.f,0.f,0.f,0.f,0.f,0.f,0.f}};

  for (int kk = 0; kk < 512; kk += 32) {
    v16bf a0 = make_afrag(xr0 + kk);
    v16bf a1 = make_afrag(xr1 + kk);

#pragma unroll
    for (int t = 0; t < 4; ++t) {
      v16bf bf = *(const v16bf*)(wb + (size_t)t * 16 * 512 + kk);
      acc0[t] = __builtin_amdgcn_wmma_f32_16x16x32_bf16(
          false, a0, false, bf, (short)0, acc0[t], false, false);
      acc1[t] = __builtin_amdgcn_wmma_f32_16x16x32_bf16(
          false, a1, false, bf, (short)0, acc1[t], false, false);
    }
  }

  // C/D layout: VGPR r -> M = r + 8*khalf, N = t*16 + lrow
#pragma unroll
  for (int t = 0; t < 4; ++t) {
    const int col = t * 16 + lrow;
#pragma unroll
    for (int r = 0; r < 8; ++r) {
      int ma = m0 + khalf * 8 + r;
      int mb = m0 + 16 + khalf * 8 + r;
      if (ma < n) h1[(size_t)ma * 64 + col] = acc0[t][r];
      if (mb < n) h1[(size_t)mb * 64 + col] = acc1[t][r];
    }
  }
}

// ---------------- per-node attention halves, layer 1: al[n,h] = sum_d h1[n,h,d]*a[h,d] ----------------
__global__ void k_prep1(const float* __restrict__ h1,
                        const float* __restrict__ a1s, const float* __restrict__ a1d,
                        float* __restrict__ al1s, float* __restrict__ al1d, int n) {
  int tid = blockIdx.x * blockDim.x + threadIdx.x;
  if (tid >= n * 8) return;
  int node = tid >> 3, hh = tid & 7;
  const float* hp = h1 + (size_t)node * 64 + hh * 8;
  const float* as = a1s + hh * 8;
  const float* ad = a1d + hh * 8;
  float ss = 0.f, dd = 0.f;
#pragma unroll
  for (int i = 0; i < 8; ++i) { float v = hp[i]; ss += v * as[i]; dd += v * ad[i]; }
  al1s[tid] = ss; al1d[tid] = dd;
}

// ---------------- init: agg=0, z=0, m=key(-inf) ----------------
__global__ void k_init(float* __restrict__ agg, float* __restrict__ z,
                       unsigned* __restrict__ m, int nagg, int nz) {
  int tid = blockIdx.x * blockDim.x + threadIdx.x;
  if (tid < nagg) agg[tid] = 0.f;
  if (tid < nz) { z[tid] = 0.f; m[tid] = KEY_NEG_INF; }
}

// ---------------- layer-1 edge passes (8 heads) ----------------
__global__ void k_e1_max(const long long* __restrict__ ei,
                         const float* __restrict__ al1s, const float* __restrict__ al1d,
                         unsigned* __restrict__ m1, int e, int n) {
  int tid = blockIdx.x * blockDim.x + threadIdx.x;
  if (tid >= e + n) return;
  int s, d;
  if (tid < e) { s = (int)ei[tid]; d = (int)ei[(size_t)e + tid]; } else { s = d = tid - e; }
  const float* ps = al1s + (size_t)s * 8;
  const float* pd = al1d + (size_t)d * 8;
#pragma unroll
  for (int h = 0; h < 8; ++h) {
    float t = lrelu(ps[h] + pd[h]);
    atomicMax(&m1[(size_t)d * 8 + h], f2key(t));
  }
}

__global__ void k_e1_sum(const long long* __restrict__ ei,
                         const float* __restrict__ al1s, const float* __restrict__ al1d,
                         const unsigned* __restrict__ m1, float* __restrict__ z1,
                         int e, int n) {
  int tid = blockIdx.x * blockDim.x + threadIdx.x;
  if (tid >= e + n) return;
  int s, d;
  if (tid < e) { s = (int)ei[tid]; d = (int)ei[(size_t)e + tid]; } else { s = d = tid - e; }
  const float* ps = al1s + (size_t)s * 8;
  const float* pd = al1d + (size_t)d * 8;
#pragma unroll
  for (int h = 0; h < 8; ++h) {
    float t = lrelu(ps[h] + pd[h]);
    float p = expf(t - key2f(m1[(size_t)d * 8 + h]));
    atomicAdd(&z1[(size_t)d * 8 + h], p);
  }
}

__global__ void k_e1_agg(const long long* __restrict__ ei,
                         const float* __restrict__ al1s, const float* __restrict__ al1d,
                         const unsigned* __restrict__ m1, const float* __restrict__ z1,
                         const float* __restrict__ h1, float* __restrict__ agg1,
                         int e, int n) {
  int tid = blockIdx.x * blockDim.x + threadIdx.x;
  if (tid >= (e + n) * 8) return;
  int eh = tid >> 3, h = tid & 7;
  int s, d;
  if (eh < e) { s = (int)ei[eh]; d = (int)ei[(size_t)e + eh]; } else { s = d = eh - e; }
  float t = lrelu(al1s[(size_t)s * 8 + h] + al1d[(size_t)d * 8 + h]);
  float p = expf(t - key2f(m1[(size_t)d * 8 + h]));
  float alpha = p / z1[(size_t)d * 8 + h];
  const float* hs = h1 + (size_t)s * 64 + h * 8;
  float* od = agg1 + (size_t)d * 64 + h * 8;
#pragma unroll
  for (int i = 0; i < 8; ++i) atomicAdd(&od[i], alpha * hs[i]);
}

// ---------------- h2 = ELU(agg1 + b1) ----------------
__global__ void k_elu(const float* __restrict__ agg1, const float* __restrict__ b1,
                      float* __restrict__ h2, int n) {
  int tid = blockIdx.x * blockDim.x + threadIdx.x;
  if (tid >= n * 64) return;
  float v = agg1[tid] + b1[tid & 63];
  h2[tid] = v > 0.f ? v : (expf(v) - 1.f);
}

// ---------------- layer-2 linear + attention halves (tiny GEMM, W2 in LDS) ----------------
__global__ void k_l2lin(const float* __restrict__ h2, const float* __restrict__ W2,
                        const float* __restrict__ a2s, const float* __restrict__ a2d,
                        float* __restrict__ g2, float* __restrict__ al2s,
                        float* __restrict__ al2d, int n) {
  __shared__ float sW[64 * 7];
  __shared__ float sas[7], sad[7];
  for (int i = threadIdx.x; i < 448; i += blockDim.x) sW[i] = W2[i];
  if (threadIdx.x < 7) { sas[threadIdx.x] = a2s[threadIdx.x]; sad[threadIdx.x] = a2d[threadIdx.x]; }
  __syncthreads();
  int node = blockIdx.x * blockDim.x + threadIdx.x;
  if (node >= n) return;
  const float* hp = h2 + (size_t)node * 64;
  float g[7];
#pragma unroll
  for (int c = 0; c < 7; ++c) g[c] = 0.f;
  for (int k = 0; k < 64; ++k) {
    float hv = hp[k];
#pragma unroll
    for (int c = 0; c < 7; ++c) g[c] += hv * sW[k * 7 + c];
  }
  float ss = 0.f, dd = 0.f;
#pragma unroll
  for (int c = 0; c < 7; ++c) {
    g2[(size_t)node * 8 + c] = g[c];
    ss += g[c] * sas[c];
    dd += g[c] * sad[c];
  }
  g2[(size_t)node * 8 + 7] = 0.f;
  al2s[node] = ss; al2d[node] = dd;
}

// ---------------- layer-2 edge passes (1 head, 7 dims) ----------------
__global__ void k_e2_max(const long long* __restrict__ ei,
                         const float* __restrict__ al2s, const float* __restrict__ al2d,
                         unsigned* __restrict__ m2, int e, int n) {
  int tid = blockIdx.x * blockDim.x + threadIdx.x;
  if (tid >= e + n) return;
  int s, d;
  if (tid < e) { s = (int)ei[tid]; d = (int)ei[(size_t)e + tid]; } else { s = d = tid - e; }
  atomicMax(&m2[d], f2key(lrelu(al2s[s] + al2d[d])));
}

__global__ void k_e2_sum(const long long* __restrict__ ei,
                         const float* __restrict__ al2s, const float* __restrict__ al2d,
                         const unsigned* __restrict__ m2, float* __restrict__ z2,
                         int e, int n) {
  int tid = blockIdx.x * blockDim.x + threadIdx.x;
  if (tid >= e + n) return;
  int s, d;
  if (tid < e) { s = (int)ei[tid]; d = (int)ei[(size_t)e + tid]; } else { s = d = tid - e; }
  float p = expf(lrelu(al2s[s] + al2d[d]) - key2f(m2[d]));
  atomicAdd(&z2[d], p);
}

__global__ void k_e2_agg(const long long* __restrict__ ei,
                         const float* __restrict__ al2s, const float* __restrict__ al2d,
                         const unsigned* __restrict__ m2, const float* __restrict__ z2,
                         const float* __restrict__ g2, float* __restrict__ agg2,
                         int e, int n) {
  int tid = blockIdx.x * blockDim.x + threadIdx.x;
  if (tid >= e + n) return;
  int s, d;
  if (tid < e) { s = (int)ei[tid]; d = (int)ei[(size_t)e + tid]; } else { s = d = tid - e; }
  float p = expf(lrelu(al2s[s] + al2d[d]) - key2f(m2[d]));
  float alpha = p / z2[d];
  const float* gs = g2 + (size_t)s * 8;
  float* od = agg2 + (size_t)d * 8;
#pragma unroll
  for (int c = 0; c < 7; ++c) atomicAdd(&od[c], alpha * gs[c]);
}

// ---------------- out = log_softmax(agg2 + b2) ----------------
__global__ void k_final(const float* __restrict__ agg2, const float* __restrict__ b2,
                        float* __restrict__ out, int n) {
  int node = blockIdx.x * blockDim.x + threadIdx.x;
  if (node >= n) return;
  float v[7];
  float mx = -INFINITY;
#pragma unroll
  for (int c = 0; c < 7; ++c) { v[c] = agg2[(size_t)node * 8 + c] + b2[c]; mx = fmaxf(mx, v[c]); }
  float sum = 0.f;
#pragma unroll
  for (int c = 0; c < 7; ++c) sum += expf(v[c] - mx);
  float lse = mx + logf(sum);
#pragma unroll
  for (int c = 0; c < 7; ++c) out[(size_t)node * 7 + c] = v[c] - lse;
}

// ---------------- host launcher ----------------
static inline size_t align256(size_t v) { return (v + 255) & ~(size_t)255; }

extern "C" void kernel_launch(void* const* d_in, const int* in_sizes, int n_in,
                              void* d_out, int out_size, void* d_ws, size_t ws_size,
                              hipStream_t stream) {
  const float*     x   = (const float*)d_in[0];
  const long long* ei  = (const long long*)d_in[1];   // int64 (2,E)
  const float*     W1  = (const float*)d_in[2];       // (512,64)
  const float*     a1s = (const float*)d_in[3];       // (8,8)
  const float*     a1d = (const float*)d_in[4];
  const float*     b1  = (const float*)d_in[5];       // (64,)
  const float*     W2  = (const float*)d_in[6];       // (64,7)
  const float*     a2s = (const float*)d_in[7];       // (1,7)
  const float*     a2d = (const float*)d_in[8];
  const float*     b2  = (const float*)d_in[9];       // (7,)
  float*           out = (float*)d_out;

  const int n = in_sizes[0] / 512;
  const int e = in_sizes[1] / 2;

  // workspace carve
  char* p = (char*)d_ws;
  size_t off = 0;
  auto carve = [&](size_t bytes) { void* r = p + off; off = align256(off + bytes); return r; };
  __bf16*   w1t  = (__bf16*)  carve((size_t)512 * 64 * 2);
  float*    h1   = (float*)   carve((size_t)n * 64 * 4);
  float*    al1s = (float*)   carve((size_t)n * 8 * 4);
  float*    al1d = (float*)   carve((size_t)n * 8 * 4);
  unsigned* m1   = (unsigned*)carve((size_t)n * 8 * 4);
  float*    z1   = (float*)   carve((size_t)n * 8 * 4);
  float*    agg1 = (float*)   carve((size_t)n * 64 * 4);
  float*    h2   = (float*)   carve((size_t)n * 64 * 4);
  float*    g2   = (float*)   carve((size_t)n * 8 * 4);
  float*    al2s = (float*)   carve((size_t)n * 4);
  float*    al2d = (float*)   carve((size_t)n * 4);
  unsigned* m2   = (unsigned*)carve((size_t)n * 4);
  float*    z2   = (float*)   carve((size_t)n * 4);
  float*    agg2 = (float*)   carve((size_t)n * 8 * 4);
  (void)ws_size;

  const int TB = 256;
  auto blocks = [&](long long work) { return (unsigned)((work + TB - 1) / TB); };

  // 1) W1 -> bf16 transposed
  k_conv_w1<<<blocks(512 * 64), TB, 0, stream>>>(W1, w1t);

  // 2) GEMM1 (WMMA bf16), 32 rows per wave
  {
    int rowTiles = (n + 31) / 32;
    int grid = (rowTiles + 7) / 8;
    k_gemm1<<<grid, TB, 0, stream>>>(x, w1t, h1, n);
  }

  // 3) per-node attention halves, layer 1
  k_prep1<<<blocks((long long)n * 8), TB, 0, stream>>>(h1, a1s, a1d, al1s, al1d, n);

  // 4) init layer-1 accumulators
  k_init<<<blocks((long long)n * 64), TB, 0, stream>>>(agg1, z1, m1, n * 64, n * 8);

  // 5-7) layer-1 edge passes
  k_e1_max<<<blocks((long long)e + n), TB, 0, stream>>>(ei, al1s, al1d, m1, e, n);
  k_e1_sum<<<blocks((long long)e + n), TB, 0, stream>>>(ei, al1s, al1d, m1, z1, e, n);
  k_e1_agg<<<blocks(((long long)e + n) * 8), TB, 0, stream>>>(ei, al1s, al1d, m1, z1, h1, agg1, e, n);

  // 8) ELU(agg1 + b1)
  k_elu<<<blocks((long long)n * 64), TB, 0, stream>>>(agg1, b1, h2, n);

  // 9) layer-2 linear + attention halves
  k_l2lin<<<blocks(n), TB, 0, stream>>>(h2, W2, a2s, a2d, g2, al2s, al2d, n);

  // 10) init layer-2 accumulators
  k_init<<<blocks((long long)n * 8), TB, 0, stream>>>(agg2, z2, m2, n * 8, n);

  // 11-13) layer-2 edge passes
  k_e2_max<<<blocks((long long)e + n), TB, 0, stream>>>(ei, al2s, al2d, m2, e, n);
  k_e2_sum<<<blocks((long long)e + n), TB, 0, stream>>>(ei, al2s, al2d, m2, z2, e, n);
  k_e2_agg<<<blocks((long long)e + n), TB, 0, stream>>>(ei, al2s, al2d, m2, z2, g2, agg2, e, n);

  // 14) log_softmax
  k_final<<<blocks(n), TB, 0, stream>>>(agg2, b2, out, n);
  (void)out_size; (void)n_in;
}
